// DownSampleCarve_52913997087003
// MI455X (gfx1250) — matrix operand order repaired and checked by reading, not compiled, not run on hardware
//
#include <hip/hip_runtime.h>
#include <stdint.h>

#define B_   2
#define CIN_ 128
#define N_   4096
#define H_   4
#define D_   32
#define M_   1024

typedef __attribute__((ext_vector_type(16))) __bf16 v16bf;
typedef __attribute__((ext_vector_type(8)))  float  v8f;

union ABu { uint32_t u[8]; v16bf v; };

__device__ __forceinline__ uint16_t f2bf(float f) {
  union { float f; uint32_t u; } x; x.f = f;
  uint32_t r = x.u + 0x7FFFu + ((x.u >> 16) & 1u);   // round-to-nearest-even
  return (uint16_t)(r >> 16);
}

// ---------------------------------------------------------------------------
// K1: q/k/v = W @ x (1x1 conv), packed to bf16 in WMMA-friendly layouts:
//   qt, kt : [B,H,N,D]  (row n contiguous over d -> A/B per-lane 32-bit pairs)
//   vh     : [B,H,D,N]  (row d contiguous over n -> AV B-operand pairs)
// ---------------------------------------------------------------------------
__global__ __launch_bounds__(256) void dsc_qkv_pack(
    const float* __restrict__ x,  const float* __restrict__ qw,
    const float* __restrict__ kw, const float* __restrict__ vw,
    uint16_t* __restrict__ qt, uint16_t* __restrict__ kt,
    uint16_t* __restrict__ vh)
{
  __shared__ float xs[CIN_ * 64];
  const int ntiles = N_ / 64;
  int blk = blockIdx.x;
  int b   = blk / (3 * ntiles);
  int r   = blk % (3 * ntiles);
  int mat = r / ntiles;
  int n0  = (r % ntiles) * 64;
  const float* w = (mat == 0) ? qw : (mat == 1) ? kw : vw;
  int tid = threadIdx.x;

  for (int t = tid; t < CIN_ * 64; t += 256) {
    int i = t >> 6, n = t & 63;
    xs[t] = x[((size_t)b * CIN_ + i) * N_ + n0 + n];
  }
  __syncthreads();

  int n  = tid & 63;
  int og = tid >> 6;
  for (int o = og; o < CIN_; o += 4) {
    float acc = 0.f;
    #pragma unroll 8
    for (int i = 0; i < CIN_; ++i)
      acc = fmaf(w[o * CIN_ + i], xs[i * 64 + n], acc);
    int h = o >> 5, d = o & 31;
    int bh = b * H_ + h;
    uint16_t bv = f2bf(acc);
    if (mat == 2) {
      vh[((size_t)bh * D_ + d) * N_ + n0 + n] = bv;
    } else {
      uint16_t* dst = (mat == 0) ? qt : kt;
      dst[((size_t)bh * N_ + (size_t)(n0 + n)) * D_ + d] = bv;
    }
  }
}

// ---------------------------------------------------------------------------
// K2: one wave per (b,h, 16-row tile). Streams over all N columns 3 times:
//   sweep1: row max of energy (WMMA bf16, K=D=32 in a single wmma per tile)
//   sweep2: row sum of exp
//   sweep3: p = softmax row values -> column-sum atomics into score,
//           bf16 relayout via LDS, P @ V^T via WMMA -> vout[b,h,n,0:32] (f32)
// ---------------------------------------------------------------------------
__global__ __launch_bounds__(32) void dsc_attn(
    const uint16_t* __restrict__ qt, const uint16_t* __restrict__ kt,
    const uint16_t* __restrict__ vh, float* __restrict__ vout,
    float* __restrict__ score)
{
  __shared__ uint16_t pl[16 * 32];           // one P tile (bf16), per-wave
  const float scale = 0.17677669529663689f;  // 1/sqrt(D)

  int wid  = blockIdx.x;
  int bh   = wid / (N_ / 16);
  int m0   = (wid % (N_ / 16)) * 16;
  int lane = threadIdx.x;
  int half = lane >> 4, lm = lane & 15;

  // A operand: 16 rows of Q^T, K-dim = D = 32 (contiguous per row)
  ABu a;
  const uint16_t* qrow = qt + ((size_t)bh * N_ + m0 + lm) * D_;
  #pragma unroll
  for (int j = 0; j < 8; ++j) {
    int kb = 16 * half + 2 * j;   // contiguous K layout (matches B)
    a.u[j] = *(const uint32_t*)(qrow + kb);
  }

  const uint16_t* kbase = kt + (size_t)bh * N_ * D_;
  const uint16_t* vbase = vh + (size_t)bh * D_ * N_;

  float rmax[8], rsum[8], rinv[8];
  #pragma unroll
  for (int j = 0; j < 8; ++j) { rmax[j] = -3.4e38f; rsum[j] = 0.f; }

  // ---- sweep 1: row max (raw; scale applied after since scale > 0) ----
  for (int n0 = 0; n0 < N_; n0 += 16) {
    ABu bb;
    const uint16_t* krow = kbase + (size_t)(n0 + lm) * D_;
    #pragma unroll
    for (int j = 0; j < 8; ++j)
      bb.u[j] = *(const uint32_t*)(krow + 16 * half + 2 * j);  // contiguous K layout (matches A)
    v8f c = {};
    c = __builtin_amdgcn_wmma_f32_16x16x32_bf16(false, a.v, false, bb.v,
                                                (short)0, c, false, false);
    #pragma unroll
    for (int j = 0; j < 8; ++j) rmax[j] = fmaxf(rmax[j], c[j]);
  }
  #pragma unroll
  for (int j = 0; j < 8; ++j) {
    #pragma unroll
    for (int off = 1; off < 16; off <<= 1)
      rmax[j] = fmaxf(rmax[j], __shfl_xor(rmax[j], off, 32));
    rmax[j] *= scale;
  }

  // ---- sweep 2: row sum of exp ----
  for (int n0 = 0; n0 < N_; n0 += 16) {
    ABu bb;
    const uint16_t* krow = kbase + (size_t)(n0 + lm) * D_;
    #pragma unroll
    for (int j = 0; j < 8; ++j)
      bb.u[j] = *(const uint32_t*)(krow + 16 * half + 2 * j);
    v8f c = {};
    c = __builtin_amdgcn_wmma_f32_16x16x32_bf16(false, a.v, false, bb.v,
                                                (short)0, c, false, false);
    #pragma unroll
    for (int j = 0; j < 8; ++j) rsum[j] += __expf(c[j] * scale - rmax[j]);
  }
  #pragma unroll
  for (int j = 0; j < 8; ++j) {
    #pragma unroll
    for (int off = 1; off < 16; off <<= 1)
      rsum[j] += __shfl_xor(rsum[j], off, 32);
    rinv[j] = 1.0f / rsum[j];
  }

  // ---- sweep 3: p -> score atomics + P @ V^T accumulation ----
  v8f acc0 = {}, acc1 = {};
  for (int n0 = 0; n0 < N_; n0 += 32) {
    ABu b0, b1;
    const uint16_t* kr0 = kbase + (size_t)(n0 + lm) * D_;
    const uint16_t* kr1 = kbase + (size_t)(n0 + 16 + lm) * D_;
    #pragma unroll
    for (int j = 0; j < 8; ++j) {
      b0.u[j] = *(const uint32_t*)(kr0 + 16 * half + 2 * j);
      b1.u[j] = *(const uint32_t*)(kr1 + 16 * half + 2 * j);
    }
    v8f c0 = {}, c1 = {};
    c0 = __builtin_amdgcn_wmma_f32_16x16x32_bf16(false, a.v, false, b0.v,
                                                 (short)0, c0, false, false);
    c1 = __builtin_amdgcn_wmma_f32_16x16x32_bf16(false, a.v, false, b1.v,
                                                 (short)0, c1, false, false);
    float sc0 = 0.f, sc1 = 0.f;
    #pragma unroll
    for (int j = 0; j < 8; ++j) {
      float p0 = __expf(c0[j] * scale - rmax[j]) * rinv[j];
      float p1 = __expf(c1[j] * scale - rmax[j]) * rinv[j];
      int rrow = j + 8 * half;                  // C-layout row of this value
      pl[rrow * 32 + lm]      = f2bf(p0);
      pl[rrow * 32 + 16 + lm] = f2bf(p1);
      sc0 += p0; sc1 += p1;
    }
    // combine the two half-tiles (rows 0-7 / 8-15) -> full column partials
    sc0 += __shfl_xor(sc0, 16, 32);
    sc1 += __shfl_xor(sc1, 16, 32);
    if (lane < 16) {
      atomicAdd(&score[bh * N_ + n0 + lm],      sc0);
      atomicAdd(&score[bh * N_ + n0 + 16 + lm], sc1);
    }
    asm volatile("s_wait_dscnt 0x0" ::: "memory");  // LDS tile visible (in-wave)
    ABu ap;
    #pragma unroll
    for (int j = 0; j < 8; ++j) {
      int kb = 16 * half + 2 * j;
      ap.u[j] = *(const uint32_t*)((const uint16_t*)pl + lm * 32 + kb);
    }
    ABu v0, v1;
    #pragma unroll
    for (int j = 0; j < 8; ++j) {
      int kb = 16 * half + 2 * j;               // K = position within chunk
      v0.u[j] = *(const uint32_t*)(vbase + (size_t)lm * N_ + n0 + kb);
      v1.u[j] = *(const uint32_t*)(vbase + (size_t)(16 + lm) * N_ + n0 + kb);
    }
    acc0 = __builtin_amdgcn_wmma_f32_16x16x32_bf16(false, ap.v, false, v0.v,
                                                   (short)0, acc0, false, false);
    acc1 = __builtin_amdgcn_wmma_f32_16x16x32_bf16(false, ap.v, false, v1.v,
                                                   (short)0, acc1, false, false);
    asm volatile("s_wait_dscnt 0x0" ::: "memory");  // protect tile reuse (WAR)
  }

  #pragma unroll
  for (int j = 0; j < 8; ++j) {
    size_t row = (size_t)bh * N_ + m0 + j + 8 * half;
    vout[row * D_ + lm]      = acc0[j];
    vout[row * D_ + 16 + lm] = acc1[j];
  }
}

// ---------------------------------------------------------------------------
// K3: rank-by-counting top-k (tie-break j<i like jax.lax.top_k), then scatter
// vout rows into channel-major x_ds / x_dropped and write idx / idx_dropped.
// One block per (b,h); score vector broadcast from LDS.
// ---------------------------------------------------------------------------
__global__ __launch_bounds__(256) void dsc_topk_scatter(
    const float* __restrict__ score, const float* __restrict__ vout,
    float* __restrict__ x_ds, int* __restrict__ idx_out,
    float* __restrict__ x_drop, int* __restrict__ idxd_out)
{
  __shared__ float s[N_];
  int bh = blockIdx.x;
  int b = bh / H_, h = bh % H_;
  int tid = threadIdx.x;
  for (int i = tid; i < N_; i += 256) s[i] = score[bh * N_ + i];
  __syncthreads();

  const int PER = N_ / 256;   // 16 elements per thread
  float si[PER]; int rank[PER];
  #pragma unroll
  for (int t = 0; t < PER; ++t) { si[t] = s[tid + 256 * t]; rank[t] = 0; }

  for (int j = 0; j < N_; ++j) {
    float sj = s[j];          // uniform address -> LDS broadcast
    #pragma unroll
    for (int t = 0; t < PER; ++t) {
      int i = tid + 256 * t;
      rank[t] += (sj > si[t]) || (sj == si[t] && j < i);
    }
  }

  for (int t = 0; t < PER; ++t) {
    int i  = tid + 256 * t;
    int rk = rank[t];
    const float* vrow = vout + ((size_t)bh * N_ + i) * D_;
    if (rk < M_) {
      idx_out[bh * M_ + rk] = i;
      #pragma unroll 8
      for (int d = 0; d < D_; ++d)
        x_ds[((size_t)b * CIN_ + h * D_ + d) * M_ + rk] = vrow[d];
    } else {
      int dp = (N_ - 1) - rk;   // ascending-score order for dropped set
      idxd_out[bh * (N_ - M_) + dp] = i;
      #pragma unroll 8
      for (int d = 0; d < D_; ++d)
        x_drop[((size_t)b * CIN_ + h * D_ + d) * (N_ - M_) + dp] = vrow[d];
    }
  }
}

// ---------------------------------------------------------------------------
extern "C" void kernel_launch(void* const* d_in, const int* in_sizes, int n_in,
                              void* d_out, int out_size, void* d_ws, size_t ws_size,
                              hipStream_t stream) {
  (void)in_sizes; (void)n_in; (void)out_size; (void)ws_size;
  const float* x  = (const float*)d_in[0];
  const float* qw = (const float*)d_in[1];
  const float* kw = (const float*)d_in[2];
  const float* vw = (const float*)d_in[3];

  // Workspace layout (~10.1 MB total):
  //   qt  bf16 [B,H,N,D]  2 MB
  //   kt  bf16 [B,H,N,D]  2 MB
  //   vh  bf16 [B,H,D,N]  2 MB
  //   vout f32 [B,H,N,D]  4 MB
  //   score f32 [B,H,N]   128 KB
  char* ws = (char*)d_ws;
  const size_t E = (size_t)B_ * CIN_ * N_;   // 1,048,576 elements
  uint16_t* qt   = (uint16_t*)(ws);
  uint16_t* kt   = (uint16_t*)(ws + E * 2);
  uint16_t* vh   = (uint16_t*)(ws + E * 4);
  float*    vout = (float*)   (ws + E * 6);
  float*    scr  = (float*)   (ws + E * 6 + E * 4);

  hipMemsetAsync(scr, 0, (size_t)B_ * H_ * N_ * sizeof(float), stream);

  dsc_qkv_pack<<<B_ * 3 * (N_ / 64), 256, 0, stream>>>(x, qw, kw, vw, qt, kt, vh);
  dsc_attn<<<B_ * H_ * (N_ / 16), 32, 0, stream>>>(qt, kt, vh, vout, scr);

  // Outputs concatenated flat: x_ds | idx | x_dropped | idx_dropped
  float* out    = (float*)d_out;
  float* x_ds   = out;                                     // B*128*M
  int*   idx_o  = (int*)(out + (size_t)B_ * CIN_ * M_);    // B*H*M
  float* x_drop = out + (size_t)B_ * CIN_ * M_ + (size_t)B_ * H_ * M_;
  int*   idxd_o = (int*)(x_drop + (size_t)B_ * CIN_ * (N_ - M_));

  dsc_topk_scatter<<<B_ * H_, 256, 0, stream>>>(scr, vout, x_ds, idx_o, x_drop, idxd_o);
}
